// Decoder_66202625901216
// MI455X (gfx1250) — matrix operand order repaired and checked by reading, not compiled, not run on hardware
//
#include <hip/hip_runtime.h>
#include <hip/hip_bf16.h>
#include <math.h>

#define VOCAB 32000
#define EMB   1024
#define HID   1024
#define NLAYERS 2
#define MAXLEN 48
#define BATCH 64
#define GATES (4*HID)

typedef __attribute__((ext_vector_type(16))) __bf16 bf16x16;
typedef __attribute__((ext_vector_type(8)))  float  v8f;

__device__ __forceinline__ unsigned short f2bf(float f) {
  unsigned u = __float_as_uint(f);
  u += 0x7FFFu + ((u >> 16) & 1u);            // round-to-nearest-even
  return (unsigned short)(u >> 16);
}

// ---------------- weight / state prep ----------------

__global__ void k_f32_to_bf16(const float* __restrict__ in,
                              unsigned short* __restrict__ out, int n) {
  int i = blockIdx.x * blockDim.x + threadIdx.x;
  int stride = gridDim.x * blockDim.x;
  for (; i < n; i += stride) out[i] = f2bf(in[i]);
}

__global__ void k_init_state(const float* __restrict__ h0, const float* __restrict__ c0,
                             unsigned short* __restrict__ hbf, float* __restrict__ c) {
  int i = blockIdx.x * blockDim.x + threadIdx.x;
  if (i >= NLAYERS * BATCH * HID) return;
  hbf[i] = f2bf(h0[i]);
  c[i] = c0[i];
}

__global__ void k_init_out(float* __restrict__ out) {
  int i = blockIdx.x * blockDim.x + threadIdx.x;
  if (i == 0) out[0] = 0.0f;                  // loss accumulator
  if (i >= 1 && i <= BATCH) out[i] = 1.0f;    // result row 0 = ones
}

// ---------------- embedding lookup (f32 -> bf16) ----------------

__global__ void k_embed(const int* __restrict__ toks, const float* __restrict__ emb,
                        unsigned short* __restrict__ x, int t) {
  int i = blockIdx.x * blockDim.x + threadIdx.x;
  if (i >= BATCH * EMB) return;
  int b = i >> 10, k = i & (EMB - 1);
  int tok = toks[t * BATCH + b];
  float v = (tok == 0) ? 0.0f : emb[tok * EMB + k];
  x[i] = f2bf(v);
}

// ---------------- WMMA tile loads (CDNA5 16x16x32 bf16 layout) ----------------

// A: 16x32 tile from row-major [M][K] activations.
// lane l -> row l&15; elems 0..7: K = kbase..kbase+7, elems 8..15: K = kbase+16..+23,
// kbase = k0 + (l>=16 ? 8 : 0).
__device__ __forceinline__ bf16x16 load_a(const unsigned short* __restrict__ base,
                                          int lda, int lane, int k0) {
  const unsigned short* p = base + (lane & 15) * lda + k0 + ((lane & 16) ? 8 : 0);
  union { uint4 u[2]; bf16x16 v; } r;
  r.u[0] = *(const uint4*)p;
  r.u[1] = *(const uint4*)(p + 16);
  return r.v;
}

// B: 32x16 tile; weight stored row-major [N][K] so B[k][n] = W[n][k].
// lane l -> column n = l&15; elems 0..7: K = kbase..kbase+7, elems 8..15: K = kbase+16..+23, kbase = k0 + (l>=16 ? 8 : 0).
__device__ __forceinline__ bf16x16 load_b(const unsigned short* __restrict__ base,
                                          int ldb, int lane, int k0) {
  const unsigned short* p = base + (lane & 15) * ldb + k0 + ((lane & 16) ? 8 : 0);
  union { uint4 u[2]; bf16x16 v; } r;
  r.u[0] = *(const uint4*)p;
  r.u[1] = *(const uint4*)(p + 16);
  return r.v;
}

// ---------------- dual-input WMMA GEMM, register-blocked over M ----------------
// C[64][N] = A1[64][K] @ W1[N][K]^T (+ A2 @ W2^T) + bias1 (+ bias2)
// One wave per 16-column tile; wave holds all 4 M-tiles (M=64) so each weight
// tile is fetched from L2 exactly once and feeds 4 WMMAs. The K-loop is
// software double-buffered: iteration k issues loads for chunk k+32 into fresh
// registers, then runs the 4 WMMAs on chunk k's registers, so WMMAs never wait
// on same-iteration loads.

__device__ __forceinline__ void gemm_dual_k(const unsigned short* __restrict__ A,
                                            const unsigned short* __restrict__ w,
                                            int K, int lane,
                                            v8f& acc0, v8f& acc1, v8f& acc2, v8f& acc3) {
  const unsigned short* A0 = A + (size_t)0 * 16 * K;
  const unsigned short* A1r = A + (size_t)1 * 16 * K;
  const unsigned short* A2r = A + (size_t)2 * 16 * K;
  const unsigned short* A3r = A + (size_t)3 * 16 * K;

  bf16x16 bv  = load_b(w,   K, lane, 0);
  bf16x16 a0v = load_a(A0,  K, lane, 0);
  bf16x16 a1v = load_a(A1r, K, lane, 0);
  bf16x16 a2v = load_a(A2r, K, lane, 0);
  bf16x16 a3v = load_a(A3r, K, lane, 0);

  #pragma unroll 2
  for (int k0 = 32; k0 < K; k0 += 32) {
    bf16x16 bn  = load_b(w,   K, lane, k0);
    bf16x16 an0 = load_a(A0,  K, lane, k0);
    bf16x16 an1 = load_a(A1r, K, lane, k0);
    bf16x16 an2 = load_a(A2r, K, lane, k0);
    bf16x16 an3 = load_a(A3r, K, lane, k0);
    acc0 = __builtin_amdgcn_wmma_f32_16x16x32_bf16(false, a0v, false, bv, (short)0, acc0, false, false);
    acc1 = __builtin_amdgcn_wmma_f32_16x16x32_bf16(false, a1v, false, bv, (short)0, acc1, false, false);
    acc2 = __builtin_amdgcn_wmma_f32_16x16x32_bf16(false, a2v, false, bv, (short)0, acc2, false, false);
    acc3 = __builtin_amdgcn_wmma_f32_16x16x32_bf16(false, a3v, false, bv, (short)0, acc3, false, false);
    bv = bn; a0v = an0; a1v = an1; a2v = an2; a3v = an3;
  }
  acc0 = __builtin_amdgcn_wmma_f32_16x16x32_bf16(false, a0v, false, bv, (short)0, acc0, false, false);
  acc1 = __builtin_amdgcn_wmma_f32_16x16x32_bf16(false, a1v, false, bv, (short)0, acc1, false, false);
  acc2 = __builtin_amdgcn_wmma_f32_16x16x32_bf16(false, a2v, false, bv, (short)0, acc2, false, false);
  acc3 = __builtin_amdgcn_wmma_f32_16x16x32_bf16(false, a3v, false, bv, (short)0, acc3, false, false);
}

__global__ void k_gemm_bf16(const unsigned short* __restrict__ A1,
                            const unsigned short* __restrict__ W1,
                            const unsigned short* __restrict__ A2,
                            const unsigned short* __restrict__ W2,
                            const float* __restrict__ bias1,
                            const float* __restrict__ bias2,
                            float* __restrict__ C, int N, int K) {
  int ntile = (blockIdx.x * blockDim.x + threadIdx.x) >> 5;
  int lane = threadIdx.x & 31;
  if (ntile >= (N >> 4)) return;        // whole-wave uniform exit

  v8f acc0 = {}, acc1 = {}, acc2 = {}, acc3 = {};

  gemm_dual_k(A1, W1 + (size_t)ntile * 16 * K, K, lane, acc0, acc1, acc2, acc3);
  if (A2)
    gemm_dual_k(A2, W2 + (size_t)ntile * 16 * K, K, lane, acc0, acc1, acc2, acc3);

  int n = ntile * 16 + (lane & 15);
  float bias = (bias1 ? bias1[n] : 0.0f) + (bias2 ? bias2[n] : 0.0f);
  int rbase = (lane & 16) ? 8 : 0;      // C layout: lane l, reg r -> row r + 8*(l>=16)
  #pragma unroll
  for (int r = 0; r < 8; ++r) {
    C[(size_t)( 0 + rbase + r) * N + n] = acc0[r] + bias;
    C[(size_t)(16 + rbase + r) * N + n] = acc1[r] + bias;
    C[(size_t)(32 + rbase + r) * N + n] = acc2[r] + bias;
    C[(size_t)(48 + rbase + r) * N + n] = acc3[r] + bias;
  }
}

// ---------------- LSTM cell (elementwise) ----------------

__global__ void k_lstm_cell(const float* __restrict__ gates,
                            float* __restrict__ c,               // [BATCH][HID], this layer
                            unsigned short* __restrict__ hbf) {  // [BATCH][HID], this layer
  int i = blockIdx.x * blockDim.x + threadIdx.x;
  if (i >= BATCH * HID) return;
  int b = i >> 10, j = i & (HID - 1);
  const float* g = gates + (size_t)b * GATES;
  float gi = g[j], gf = g[HID + j], gg = g[2 * HID + j], go = g[3 * HID + j];
  float si = 1.0f / (1.0f + expf(-gi));
  float sf = 1.0f / (1.0f + expf(-gf));
  float so = 1.0f / (1.0f + expf(-go));
  float cn = sf * c[i] + si * tanhf(gg);
  float hn = so * tanhf(cn);
  c[i] = cn;
  hbf[i] = f2bf(hn);
}

// ---------------- per-row softmax / NLL / argmax ----------------

__global__ void k_softmax_nll(const float* __restrict__ logits,
                              const int* __restrict__ toks,
                              float* __restrict__ nllbuf,   // [0..63]=nll*mask, [64..127]=mask
                              float* __restrict__ out, int t) {
  __shared__ float smax[256]; __shared__ int sidx[256]; __shared__ float ssum[256];
  int b = blockIdx.x, tid = threadIdx.x;
  const float* row = logits + (size_t)b * VOCAB;

  float m = -INFINITY; int mi = 0;
  for (int j = tid; j < VOCAB; j += 256) {
    float v = row[j];
    if (v > m) { m = v; mi = j; }
  }
  smax[tid] = m; sidx[tid] = mi;
  __syncthreads();
  for (int s = 128; s > 0; s >>= 1) {
    if (tid < s) {
      float om = smax[tid + s]; int oi = sidx[tid + s];
      if (om > smax[tid] || (om == smax[tid] && oi < sidx[tid])) { smax[tid] = om; sidx[tid] = oi; }
    }
    __syncthreads();
  }
  float rmax = smax[0]; int amax = sidx[0];

  float s = 0.0f;
  for (int j = tid; j < VOCAB; j += 256) s += expf(row[j] - rmax);
  ssum[tid] = s;
  __syncthreads();
  for (int st = 128; st > 0; st >>= 1) { if (tid < st) ssum[tid] += ssum[tid + st]; __syncthreads(); }

  if (tid == 0) {
    int tgt = toks[(t + 1) * BATCH + b];
    float lse = logf(ssum[0]) + rmax;
    float nll = lse - row[tgt];
    float mask = (tgt != 0) ? 1.0f : 0.0f;
    nllbuf[b] = nll * mask;
    nllbuf[BATCH + b] = mask;
    out[1 + (t + 1) * BATCH + b] = (float)amax;   // result[t+1][b]
  }
}

__global__ void k_ce_reduce(const float* __restrict__ nllbuf, float* __restrict__ out) {
  __shared__ float sn[64]; __shared__ float sm[64];
  int tid = threadIdx.x;
  sn[tid] = nllbuf[tid]; sm[tid] = nllbuf[BATCH + tid];
  __syncthreads();
  for (int s = 32; s > 0; s >>= 1) {
    if (tid < s) { sn[tid] += sn[tid + s]; sm[tid] += sm[tid + s]; }
    __syncthreads();
  }
  if (tid == 0) out[0] += sn[0] / fmaxf(sm[0], 1.0f);
}

// ---------------- host side ----------------

extern "C" void kernel_launch(void* const* d_in, const int* in_sizes, int n_in,
                              void* d_out, int out_size, void* d_ws, size_t ws_size,
                              hipStream_t stream) {
  const int*   toks  = (const int*)  d_in[0];   // output_tensor (48,64)
  // d_in[1] encoder_hidden_states, d_in[2] input_mask: unused by the reference math
  const float* h0    = (const float*)d_in[3];   // (2,64,1024)
  const float* c0    = (const float*)d_in[4];   // (2,64,1024)
  const float* emb   = (const float*)d_in[5];   // (32000,1024)
  const float* w_ih  = (const float*)d_in[6];   // (2,4096,1024)
  const float* w_hh  = (const float*)d_in[7];   // (2,4096,1024)
  const float* b_ih  = (const float*)d_in[8];   // (2,4096)
  const float* b_hh  = (const float*)d_in[9];   // (2,4096)
  const float* W_out = (const float*)d_in[10];  // (32000,1024)
  const float* b_out = (const float*)d_in[11];  // (32000,)
  float* out = (float*)d_out;                   // [loss(1)] + [result 48*64]

  char* ws = (char*)d_ws;
  size_t off = 0;
  auto alloc = [&](size_t bytes) -> char* {
    char* p = ws + off;
    off = (off + bytes + 255) & ~(size_t)255;
    return p;
  };
  unsigned short* wih_bf  = (unsigned short*)alloc((size_t)NLAYERS * GATES * EMB * 2);
  unsigned short* whh_bf  = (unsigned short*)alloc((size_t)NLAYERS * GATES * HID * 2);
  unsigned short* wout_bf = (unsigned short*)alloc((size_t)VOCAB * HID * 2);
  unsigned short* x_bf    = (unsigned short*)alloc((size_t)BATCH * EMB * 2);
  unsigned short* h_bf    = (unsigned short*)alloc((size_t)NLAYERS * BATCH * HID * 2);
  float*          c_st    = (float*)alloc((size_t)NLAYERS * BATCH * HID * 4);
  float*          gatesb  = (float*)alloc((size_t)BATCH * GATES * 4);
  float*          logits  = (float*)alloc((size_t)BATCH * VOCAB * 4);
  float*          nllbuf  = (float*)alloc((size_t)2 * BATCH * 4);

  // one-time (per launch) weight conversion to bf16: ~99 MB, L2-resident thereafter
  k_f32_to_bf16<<<2048, 256, 0, stream>>>(w_ih,  wih_bf,  NLAYERS * GATES * EMB);
  k_f32_to_bf16<<<2048, 256, 0, stream>>>(w_hh,  whh_bf,  NLAYERS * GATES * HID);
  k_f32_to_bf16<<<4096, 256, 0, stream>>>(W_out, wout_bf, VOCAB * HID);
  k_init_state<<<(NLAYERS * BATCH * HID + 255) / 256, 256, 0, stream>>>(h0, c0, h_bf, c_st);
  k_init_out<<<1, 128, 0, stream>>>(out);

  const int gate_blocks  = ((GATES / 16) + 7) / 8;   // 32  (256 ntile-waves)
  const int logit_blocks = ((VOCAB / 16) + 7) / 8;   // 250 (2000 ntile-waves)

  for (int t = 0; t < MAXLEN - 1; ++t) {
    k_embed<<<(BATCH * EMB + 255) / 256, 256, 0, stream>>>(toks, emb, x_bf, t);

    for (int l = 0; l < NLAYERS; ++l) {
      const unsigned short* in_bf = (l == 0) ? x_bf : (h_bf + 0 * BATCH * HID);
      k_gemm_bf16<<<gate_blocks, 256, 0, stream>>>(
          in_bf,                          wih_bf + (size_t)l * GATES * EMB,
          h_bf + (size_t)l * BATCH * HID, whh_bf + (size_t)l * GATES * HID,
          b_ih + (size_t)l * GATES,       b_hh + (size_t)l * GATES,
          gatesb, GATES, EMB);
      k_lstm_cell<<<(BATCH * HID + 255) / 256, 256, 0, stream>>>(
          gatesb, c_st + (size_t)l * BATCH * HID, h_bf + (size_t)l * BATCH * HID);
    }

    k_gemm_bf16<<<logit_blocks, 256, 0, stream>>>(
        h_bf + (size_t)1 * BATCH * HID, wout_bf,
        nullptr, nullptr, b_out, nullptr, logits, VOCAB, HID);

    k_softmax_nll<<<BATCH, 256, 0, stream>>>(logits, toks, nllbuf, out, t);
    k_ce_reduce<<<1, 64, 0, stream>>>(nllbuf, out);
  }
}
